// BatchTreeEncoder_4191888081356
// MI455X (gfx1250) — matrix operand order repaired and checked by reading, not compiled, not run on hardware
//
#include <hip/hip_runtime.h>
#include <hip/hip_bf16.h>

typedef __attribute__((ext_vector_type(4)))  float  v4f;
typedef __attribute__((ext_vector_type(8)))  float  v8f;
typedef __attribute__((ext_vector_type(8)))  __bf16 v8bf;
typedef __attribute__((ext_vector_type(16))) __bf16 v16bf;

#define TDEPTH 10
#define NNODE  1023   // real nodes
#define NPAD   1024   // padded (node 1023 duplicates leaf 1022, cnt forced to 1)
#define EMB    128
#define ENC    128
#define BSZ    256

static __device__ __forceinline__ float bf2f(unsigned v) {
  union { unsigned u; float f; } x; x.u = v << 16; return x.f;
}
static __device__ __forceinline__ unsigned f2bf(float f) {
  union { unsigned u; float f; } x; x.f = f;
  unsigned r = x.u + 0x7FFFu + ((x.u >> 16) & 1u);
  return r >> 16;
}
// packed 2xbf16: a + b + c (f32 internally, RNE repack)
static __device__ __forceinline__ unsigned addpk(unsigned a, unsigned b, unsigned c) {
  float lo = bf2f(a & 0xFFFFu) + bf2f(b & 0xFFFFu) + bf2f(c & 0xFFFFu);
  float hi = bf2f(a >> 16)     + bf2f(b >> 16)     + bf2f(c >> 16);
  return (f2bf(lo) & 0xFFFFu) | (f2bf(hi) << 16);
}

// One block per batch element. 256 threads = 8 waves (wave32).
// LDS: S (1024x128 bf16 = 256KB) + Wc (128x128 bf16 = 32KB) + cntf (4KB) = 292KB.
__global__ void __launch_bounds__(256, 1)
batch_tree_encoder_kernel(const int* __restrict__ tok,
                          const float* __restrict__ emb,
                          const float* __restrict__ Wc,
                          const float* __restrict__ bc,
                          float* __restrict__ out) {
  extern __shared__ unsigned char smem[];
  unsigned* Sw  = (unsigned*)smem;                               // 1024*64 u32
  unsigned* Ww  = (unsigned*)(smem + NPAD * EMB * 2);            // 128*64  u32
  float*    Cf  = (float*)   (smem + NPAD * EMB * 2 + ENC * EMB * 2); // 1024 f32
  const __bf16* Sb = (const __bf16*)Sw;
  const __bf16* Wb = (const __bf16*)Ww;

  const int tid  = threadIdx.x;
  const int b    = blockIdx.x;
  const int wave = tid >> 5;
  const int lane = tid & 31;
  const int r    = lane & 15;
  const int half = lane >> 4;

  // ---- Phase 0a: W_c (row o, col e) -> LDS bf16 ----------------------------
  for (int i = tid; i < (ENC * EMB) / 4; i += 256) {
    float4 w = ((const float4*)Wc)[i];
    unsigned lo = (f2bf(w.x) & 0xFFFFu) | (f2bf(w.y) << 16);
    unsigned hi = (f2bf(w.z) & 0xFFFFu) | (f2bf(w.w) << 16);
    ((uint2*)Ww)[i] = make_uint2(lo, hi);
  }

  // ---- Phase 0b: subtree-count table cntf[node] = 2^(10-level)-1 -----------
  // (pad node 1023 gets cnt=1 so it exactly replicates leaf 1022's h)
  for (int i = tid; i < NPAD; i += 256) {
    int lvl = 31 - __clz(i + 1);
    Cf[i] = (i == NNODE) ? 1.0f : (float)((1 << (TDEPTH - lvl)) - 1);
  }

  // ---- Phase 1: gather embeddings -> LDS bf16 ------------------------------
  // wave per node; lane loads 4 consecutive floats (16B), stores 8B packed bf16
  for (int n = wave; n < NPAD; n += 8) {
    int nn  = (n < NNODE) ? n : (NNODE - 1);      // pad node dups leaf 1022
    int row = tok[b * NNODE + nn];
    float4 e = ((const float4*)(emb + row * EMB))[lane];
    unsigned lo = (f2bf(e.x) & 0xFFFFu) | (f2bf(e.y) << 16);
    unsigned hi = (f2bf(e.z) & 0xFFFFu) | (f2bf(e.w) << 16);
    ((uint2*)(Sw + n * 64))[lane] = make_uint2(lo, hi);
  }
  __syncthreads();

  // ---- Phase 2: bottom-up subtree sum of embeddings in LDS -----------------
  for (int l = TDEPTH - 2; l >= 0; --l) {
    const int s   = (1 << l) - 1;
    const int ngr = (1 << l) * 16;                // 16 uint4-groups per node
    for (int g = tid; g < ngr; g += 256) {
      int node = s + (g >> 4);
      int q    = (g & 15) << 2;                   // u32 word offset in row
      uint4 a   = *(const uint4*)(Sw + node * 64 + q);
      uint4 lft = *(const uint4*)(Sw + (2 * node + 1) * 64 + q);
      uint4 rgt = *(const uint4*)(Sw + (2 * node + 2) * 64 + q);
      uint4 res;
      res.x = addpk(a.x, lft.x, rgt.x);
      res.y = addpk(a.y, lft.y, rgt.y);
      res.z = addpk(a.z, lft.z, rgt.z);
      res.w = addpk(a.w, lft.w, rgt.w);
      *(uint4*)(Sw + node * 64 + q) = res;
    }
    __syncthreads();
  }

  // ---- Phase 3: WMMA GEMM (S @ W_c^T) + bias + running max -----------------
  // wave owns channel tile [wave*16, wave*16+16); iterates 64 node tiles.
  const int   ch  = wave * 16 + r;
  const float bcv = bc[ch];

  // B fragments: column n = r holds W_c[ch][k]; per-lane K = k0 + {8*half+e, 16+8*half+e}.
  // Invariant across node tiles -> hoisted (loaded once, 32 VGPRs).
  v16bf Bf[4];
#pragma unroll
  for (int ks = 0; ks < 4; ++ks) {
    const __bf16* wrow = Wb + ch * EMB + ks * 32 + 8 * half;
    v8bf lo = *(const v8bf*)(wrow);
    v8bf hi = *(const v8bf*)(wrow + 16);
    Bf[ks] = __builtin_shufflevector(lo, hi, 0, 1, 2, 3, 4, 5, 6, 7,
                                     8, 9, 10, 11, 12, 13, 14, 15);
  }

  v8f mx = { -3.402823466e38f, -3.402823466e38f, -3.402823466e38f,
             -3.402823466e38f, -3.402823466e38f, -3.402823466e38f,
             -3.402823466e38f, -3.402823466e38f };

  // A layout (16-bit 16x32): lane row M=r; K = ks*32 + {8*half+e, 16+8*half+e}
  const __bf16* SbL = Sb + r * EMB + 8 * half;    // lane-fixed base

#define LOAD_A(dst, tile)                                                     \
  {                                                                           \
    const __bf16* sp_ = SbL + ((tile) * 16) * EMB;                            \
    _Pragma("unroll") for (int ks_ = 0; ks_ < 4; ++ks_) {                     \
      v8bf lo_ = *(const v8bf*)(sp_ + ks_ * 32);                              \
      v8bf hi_ = *(const v8bf*)(sp_ + ks_ * 32 + 16);                         \
      dst[ks_] = __builtin_shufflevector(lo_, hi_, 0, 1, 2, 3, 4, 5, 6, 7,    \
                                         8, 9, 10, 11, 12, 13, 14, 15);       \
    }                                                                         \
  }

#define COMPUTE_TILE(afrag, tile)                                             \
  {                                                                           \
    v8f acc_ = { 0.f, 0.f, 0.f, 0.f, 0.f, 0.f, 0.f, 0.f };                    \
    _Pragma("unroll") for (int ks_ = 0; ks_ < 4; ++ks_) {                     \
      acc_ = __builtin_amdgcn_wmma_f32_16x16x32_bf16(                         \
          false, afrag[ks_], false, Bf[ks_], (short)0, acc_, false, false);   \
    }                                                                         \
    const float* cp_ = Cf + (tile) * 16 + 8 * half;                           \
    v4f c0_ = *(const v4f*)(cp_);                                             \
    v4f c1_ = *(const v4f*)(cp_ + 4);                                         \
    _Pragma("unroll") for (int v_ = 0; v_ < 4; ++v_) {                        \
      mx[v_]     = fmaxf(mx[v_],     fmaf(c0_[v_], bcv, acc_[v_]));           \
      mx[v_ + 4] = fmaxf(mx[v_ + 4], fmaf(c1_[v_], bcv, acc_[v_ + 4]));       \
    }                                                                         \
  }

  // explicit ping-pong double buffering (no register copies):
  // tile t computes from A0 while A1 prefetches t+1, and vice versa.
  v16bf A0[4], A1[4];
  LOAD_A(A0, 0);
  for (int t = 0; t < NPAD / 16; t += 2) {
    LOAD_A(A1, t + 1);                 // t+1 <= 63, always valid
    COMPUTE_TILE(A0, t);
    LOAD_A(A0, (t + 2) & 63);          // wraps to 0 on last trip (harmless)
    COMPUTE_TILE(A1, t + 1);
  }
#undef LOAD_A
#undef COMPUTE_TILE

  // reduce: 8 accumulator rows, then the complementary half-lane (same channel)
  float m = mx[0];
#pragma unroll
  for (int v = 1; v < 8; ++v) m = fmaxf(m, mx[v]);
  m = fmaxf(m, __shfl_xor(m, 16, 32));
  if (half == 0) out[b * ENC + ch] = m;
}

extern "C" void kernel_launch(void* const* d_in, const int* in_sizes, int n_in,
                              void* d_out, int out_size, void* d_ws, size_t ws_size,
                              hipStream_t stream) {
  (void)in_sizes; (void)n_in; (void)out_size; (void)d_ws; (void)ws_size;
  const int*   tok = (const int*)d_in[0];
  const float* emb = (const float*)d_in[1];
  const float* Wc  = (const float*)d_in[2];
  const float* bc  = (const float*)d_in[3];
  float*       out = (float*)d_out;

  const size_t shmem = (size_t)NPAD * EMB * 2   // S  bf16
                     + (size_t)ENC * EMB * 2    // Wc bf16
                     + (size_t)NPAD * 4;        // cntf f32   => 292KB
  hipFuncSetAttribute(reinterpret_cast<const void*>(&batch_tree_encoder_kernel),
                      hipFuncAttributeMaxDynamicSharedMemorySize, (int)shmem);
  batch_tree_encoder_kernel<<<BSZ, 256, shmem, stream>>>(tok, emb, Wc, bc, out);
}
